// SparseSelfAttention_47038481825972
// MI455X (gfx1250) — compile-verified, matrix operand order checked by reading, not run on hardware
//
#include <hip/hip_runtime.h>
#include <hip/hip_bf16.h>

typedef __attribute__((ext_vector_type(16))) _Float16 v16h;
typedef __attribute__((ext_vector_type(8)))  _Float16 v8h;
typedef __attribute__((ext_vector_type(8)))  float    v8f;

#define BATCH 16
#define SEQ   2048
#define DIM   256

// ---------------------------------------------------------------------------
// WMMA 16x16x32 f16 operand loaders (wave32, gfx1250 layout).
// A-matrix (16xK, row m striped across VGPR halves) and B-matrix (Kx16,
// column n) both map to: lane (r = lane&15 selects row/col, h = lane>>4) reads
//   elems [k0+8h .. k0+8h+7]        -> v16h[0..7]
//   elems [k0+16+8h .. k0+16+8h+7]  -> v16h[8..15]
// i.e. two contiguous 8-element chunks of a row-major row.
// ---------------------------------------------------------------------------
__device__ __forceinline__ v16h load_row_f32_as_f16(const float* __restrict__ row,
                                                    int k0, int h) {
    const float4* p0 = (const float4*)(row + k0 + 8 * h);
    const float4* p1 = (const float4*)(row + k0 + 16 + 8 * h);
    float4 f0 = p0[0], f1 = p0[1];
    float4 f2 = p1[0], f3 = p1[1];
    v16h r;
    r[0]  = (_Float16)f0.x; r[1]  = (_Float16)f0.y;
    r[2]  = (_Float16)f0.z; r[3]  = (_Float16)f0.w;
    r[4]  = (_Float16)f1.x; r[5]  = (_Float16)f1.y;
    r[6]  = (_Float16)f1.z; r[7]  = (_Float16)f1.w;
    r[8]  = (_Float16)f2.x; r[9]  = (_Float16)f2.y;
    r[10] = (_Float16)f2.z; r[11] = (_Float16)f2.w;
    r[12] = (_Float16)f3.x; r[13] = (_Float16)f3.y;
    r[14] = (_Float16)f3.z; r[15] = (_Float16)f3.w;
    return r;
}

__device__ __forceinline__ v16h load_row_f16(const _Float16* __restrict__ row,
                                             int k0, int h) {
    v8h lo = *(const v8h*)(row + k0 + 8 * h);        // 16B load
    v8h hi = *(const v8h*)(row + k0 + 16 + 8 * h);   // 16B load
    v16h r;
#pragma unroll
    for (int i = 0; i < 8; ++i) { r[i] = lo[i]; r[8 + i] = hi[i]; }
    return r;
}

// ---------------------------------------------------------------------------
// Kernel 1: scores[b] = Q[b] (NxD) x K[b]^T (DxN), written to A region.
// Wave tile 16(M)x64(N); 8 waves stacked in M -> workgroup tile 128x64.
// ---------------------------------------------------------------------------
__global__ void __launch_bounds__(256)
qk_scores_kernel(const float* __restrict__ Q, const float* __restrict__ K,
                 float* __restrict__ S) {
    const int b    = blockIdx.z;
    const int lane = threadIdx.x & 31;
    const int wave = threadIdx.x >> 5;
    const int m0   = blockIdx.y * 128 + wave * 16;
    const int n0   = blockIdx.x * 64;
    const int r    = lane & 15;
    const int h    = lane >> 4;

    const float* Qb = Q + (size_t)b * SEQ * DIM;
    const float* Kb = K + (size_t)b * SEQ * DIM;
    float*       Sb = S + (size_t)b * SEQ * SEQ;

    const float* qrow = Qb + (size_t)(m0 + r) * DIM;
    const float* krow = Kb + (size_t)(n0 + r) * DIM;

    v8f acc[4] = {};
#pragma unroll
    for (int k0 = 0; k0 < DIM; k0 += 32) {
        v16h a = load_row_f32_as_f16(qrow, k0, h);
#pragma unroll
        for (int t = 0; t < 4; ++t) {
            v16h bm = load_row_f32_as_f16(krow + (size_t)t * 16 * DIM, k0, h);
            acc[t] = __builtin_amdgcn_wmma_f32_16x16x32_f16(
                false, a, false, bm, (short)0, acc[t], false, false);
        }
    }
    // C/D layout: VGPR v of lane (r,h) holds element (M = v + 8h, N = r).
#pragma unroll
    for (int t = 0; t < 4; ++t)
#pragma unroll
        for (int v = 0; v < 8; ++v)
            Sb[(size_t)(m0 + 8 * h + v) * SEQ + n0 + t * 16 + r] = acc[t][v];
}

// ---------------------------------------------------------------------------
// Kernel 2: sparsemax per row (Michelot exact simplex-projection threshold).
// One 256-thread block per row; row cached in LDS. Writes gates in-place over
// the scores (A region), a second f32 copy to the gates region, and (if the
// workspace is large enough) an f16 shadow copy for the AV GEMM's A operand.
// ---------------------------------------------------------------------------
__device__ __forceinline__ void block_reduce2(float& a, float& b,
                                              volatile float* sa,
                                              volatile float* sb, int tid) {
#pragma unroll
    for (int off = 16; off > 0; off >>= 1) {
        a += __shfl_down(a, off, 32);
        b += __shfl_down(b, off, 32);
    }
    const int w = tid >> 5;
    if ((tid & 31) == 0) { sa[w] = a; sb[w] = b; }
    __syncthreads();
    if (tid == 0) {
        float ra = 0.f, rb = 0.f;
#pragma unroll
        for (int i = 0; i < 8; ++i) { ra += sa[i]; rb += sb[i]; }
        sa[0] = ra; sb[0] = rb;
    }
    __syncthreads();
    a = sa[0];
    b = sb[0];
    __syncthreads();
}

__global__ void __launch_bounds__(256)
sparsemax_kernel(float* __restrict__ A, float* __restrict__ G,
                 _Float16* __restrict__ GH) {
    __shared__ float z[SEQ];
    __shared__ float red_a[8];
    __shared__ float red_b[8];

    const int tid = threadIdx.x;
    const int row = blockIdx.x;
    const int b   = blockIdx.y;
    const size_t roff = ((size_t)b * SEQ + row) * SEQ;
    float* arow = A + roff;
    float* grow = G + roff;

    for (int i = tid; i < SEQ; i += 256) z[i] = arow[i];
    __syncthreads();

    // Iteration 0: full support.
    float s = 0.f, c = 0.f;
    for (int i = tid; i < SEQ; i += 256) s += z[i];
    block_reduce2(s, c, red_a, red_b, tid);
    float tau = (s - 1.0f) / (float)SEQ;
    int k_prev = SEQ;

    // Michelot: shrink support {z > tau}; support shrinks monotonically, so
    // k unchanged => support unchanged => converged (exact sparsemax tau).
    for (int it = 0; it < 64; ++it) {
        s = 0.f; c = 0.f;
        for (int i = tid; i < SEQ; i += 256) {
            float zi = z[i];
            if (zi > tau) { s += zi; c += 1.0f; }
        }
        block_reduce2(s, c, red_a, red_b, tid);
        int k = (int)c;
        if (k == k_prev || k == 0) break;
        tau = (s - 1.0f) / c;
        k_prev = k;
    }

    if (GH != nullptr) {
        _Float16* hrow = GH + roff;
        for (int i = tid; i < SEQ; i += 256) {
            float p = z[i] - tau;
            p = p > 0.f ? p : 0.f;
            arow[i] = p;
            grow[i] = p;
            hrow[i] = (_Float16)p;
        }
    } else {
        for (int i = tid; i < SEQ; i += 256) {
            float p = z[i] - tau;
            p = p > 0.f ? p : 0.f;
            arow[i] = p;
            grow[i] = p;
        }
    }
}

// ---------------------------------------------------------------------------
// Kernel 3: values[b] (SxD f32) -> VT[b] (DxS f16) via LDS tile transpose,
// so the AV GEMM's B-matrix loads become contiguous 16B f16 loads.
// ---------------------------------------------------------------------------
__global__ void __launch_bounds__(256)
transpose_values_kernel(const float* __restrict__ V, _Float16* __restrict__ VT) {
    __shared__ _Float16 tile[32][33];
    const int b  = blockIdx.z;
    const int s0 = blockIdx.x * 32;
    const int d0 = blockIdx.y * 32;
    const int tx = threadIdx.x & 31;
    const int ty = threadIdx.x >> 5;   // 0..7

    const float* Vb  = V  + (size_t)b * SEQ * DIM;
    _Float16*    VTb = VT + (size_t)b * SEQ * DIM;

#pragma unroll
    for (int j = 0; j < 32; j += 8)
        tile[ty + j][tx] = (_Float16)Vb[(size_t)(s0 + ty + j) * DIM + d0 + tx];
    __syncthreads();
#pragma unroll
    for (int j = 0; j < 32; j += 8)
        VTb[(size_t)(d0 + ty + j) * SEQ + s0 + tx] = tile[tx][ty + j];
}

// ---------------------------------------------------------------------------
// Kernel 4a (fast path): Vout[b] = gates_f16[b] (NxS) x values[b] via VT.
// Both operands are contiguous f16 rows -> pure b128 loads + wmma, no cvt.
// ---------------------------------------------------------------------------
__global__ void __launch_bounds__(256)
av_kernel_h(const _Float16* __restrict__ GH, const _Float16* __restrict__ VT,
            float* __restrict__ O) {
    const int b    = blockIdx.z;
    const int lane = threadIdx.x & 31;
    const int wave = threadIdx.x >> 5;
    const int m0   = blockIdx.y * 128 + wave * 16;
    const int d0   = blockIdx.x * 64;
    const int r    = lane & 15;
    const int h    = lane >> 4;

    const _Float16* Gb  = GH + (size_t)b * SEQ * SEQ;
    const _Float16* VTb = VT + (size_t)b * SEQ * DIM;
    float*          Ob  = O  + (size_t)b * SEQ * DIM;

    const _Float16* arow = Gb  + (size_t)(m0 + r) * SEQ;
    const _Float16* vrow = VTb + (size_t)(d0 + r) * SEQ;

    v8f acc[4] = {};
    for (int k0 = 0; k0 < SEQ; k0 += 32) {
        v16h a = load_row_f16(arow, k0, h);
#pragma unroll
        for (int t = 0; t < 4; ++t) {
            v16h bm = load_row_f16(vrow + (size_t)t * 16 * SEQ, k0, h);
            acc[t] = __builtin_amdgcn_wmma_f32_16x16x32_f16(
                false, a, false, bm, (short)0, acc[t], false, false);
        }
    }
#pragma unroll
    for (int t = 0; t < 4; ++t)
#pragma unroll
        for (int v = 0; v < 8; ++v)
            Ob[(size_t)(m0 + 8 * h + v) * DIM + d0 + t * 16 + r] = acc[t][v];
}

// ---------------------------------------------------------------------------
// Kernel 4b (fallback): A operand read as f32 from d_out, converted on the fly.
// ---------------------------------------------------------------------------
__global__ void __launch_bounds__(256)
av_kernel_f(const float* __restrict__ A, const _Float16* __restrict__ VT,
            float* __restrict__ O) {
    const int b    = blockIdx.z;
    const int lane = threadIdx.x & 31;
    const int wave = threadIdx.x >> 5;
    const int m0   = blockIdx.y * 128 + wave * 16;
    const int d0   = blockIdx.x * 64;
    const int r    = lane & 15;
    const int h    = lane >> 4;

    const float*    Ab  = A  + (size_t)b * SEQ * SEQ;
    const _Float16* VTb = VT + (size_t)b * SEQ * DIM;
    float*          Ob  = O  + (size_t)b * SEQ * DIM;

    const float*    arow = Ab  + (size_t)(m0 + r) * SEQ;
    const _Float16* vrow = VTb + (size_t)(d0 + r) * SEQ;

    v8f acc[4] = {};
    for (int k0 = 0; k0 < SEQ; k0 += 32) {
        v16h a = load_row_f32_as_f16(arow, k0, h);
#pragma unroll
        for (int t = 0; t < 4; ++t) {
            v16h bm = load_row_f16(vrow + (size_t)t * 16 * SEQ, k0, h);
            acc[t] = __builtin_amdgcn_wmma_f32_16x16x32_f16(
                false, a, false, bm, (short)0, acc[t], false, false);
        }
    }
#pragma unroll
    for (int t = 0; t < 4; ++t)
#pragma unroll
        for (int v = 0; v < 8; ++v)
            Ob[(size_t)(m0 + 8 * h + v) * DIM + d0 + t * 16 + r] = acc[t][v];
}

// ---------------------------------------------------------------------------
// Launch: d_out = [Vout (B*N*D) | A (B*N*S) | gates (B*N*S)] fp32.
// d_ws: [VT (B*D*S f16, 16.8 MB)][optional gates-f16 (B*N*S f16, 134 MB)].
// Path choice depends only on ws_size -> deterministic across calls.
// ---------------------------------------------------------------------------
extern "C" void kernel_launch(void* const* d_in, const int* in_sizes, int n_in,
                              void* d_out, int out_size, void* d_ws, size_t ws_size,
                              hipStream_t stream) {
    const float* Q = (const float*)d_in[0];
    const float* K = (const float*)d_in[1];
    const float* V = (const float*)d_in[2];

    float* out  = (float*)d_out;
    float* Vout = out;
    float* Aout = Vout + (size_t)BATCH * SEQ * DIM;
    float* Gout = Aout + (size_t)BATCH * SEQ * SEQ;

    const size_t vt_bytes = (size_t)BATCH * SEQ * DIM * sizeof(_Float16);
    const size_t gh_bytes = (size_t)BATCH * SEQ * SEQ * sizeof(_Float16);
    const bool   use_h    = (ws_size >= vt_bytes + gh_bytes);

    _Float16* VT = (_Float16*)d_ws;
    _Float16* GH = use_h ? (_Float16*)((char*)d_ws + vt_bytes) : nullptr;

    // 1) scores = Q K^T  -> A region
    dim3 g1(SEQ / 64, SEQ / 128, BATCH);
    qk_scores_kernel<<<g1, 256, 0, stream>>>(Q, K, Aout);

    // 2) sparsemax rows in-place, duplicate into gates region (+ f16 shadow)
    dim3 g2(SEQ, BATCH);
    sparsemax_kernel<<<g2, 256, 0, stream>>>(Aout, Gout, GH);

    // 3) values -> f16 transposed workspace
    dim3 g3(SEQ / 32, DIM / 32, BATCH);
    transpose_values_kernel<<<g3, 256, 0, stream>>>(V, VT);

    // 4) Vout = gates x values
    dim3 g4(DIM / 64, SEQ / 128, BATCH);
    if (use_h)
        av_kernel_h<<<g4, 256, 0, stream>>>(GH, VT, Vout);
    else
        av_kernel_f<<<g4, 256, 0, stream>>>(Aout, VT, Vout);
}
